// zSparseAttention_34746285425375
// MI455X (gfx1250) — matrix operand order repaired and checked by reading, not compiled, not run on hardware
//
#include <hip/hip_runtime.h>

typedef _Float16 v16h __attribute__((ext_vector_type(16)));
typedef float    v8f  __attribute__((ext_vector_type(8)));

#define S_LEN 4096
#define DHEAD 128
#define WIN   512
#define KSTR  136   // halves per K row in LDS (128 + 8 pad)
#define VSTR  40    // halves per Vt row in LDS (32 + 8 pad)

__device__ __forceinline__ void load_pair(const _Float16* p, v16h& dst, int vv) {
    // p is 4-byte aligned (even half offset); low address -> low 16 bits -> even K
    union { float f; _Float16 h[2]; } u;
    u.f = *reinterpret_cast<const float*>(p);
    dst[2 * vv]     = u.h[0];
    dst[2 * vv + 1] = u.h[1];
}

// One 32-key chunk, transposed formulation: S^T = K*Q^T, O^T += V^T*P^T.
// Scores are in base-2 units (log2e folded into Q scale); exp via v_exp_f32.
// Lane n (both halves) owns query row q0w+n; keys live in the VGPR index.
template <bool MASKED>
__device__ __forceinline__ void chunk_compute(
    const _Float16* __restrict__ Kb, const _Float16* __restrict__ Vb,
    const v16h* qf, v8f* oacc, float& mst, float& lst,
    int jbase, int qrow, int ln, int half) {
    // ---- S^T tiles: A = K rows (lane = key), B = Q^T (lane = own q row) ----
    float sc[2][8];
#pragma unroll
    for (int ct = 0; ct < 2; ++ct) {
        v8f acc;
#pragma unroll
        for (int e = 0; e < 8; ++e) acc[e] = 0.0f;
        const _Float16* kp = &Kb[(16 * ct + ln) * KSTR];
#pragma unroll
        for (int s = 0; s < 4; ++s) {
            v16h aK;  // A layout: K-dim d = 32s + 2vv + 16*half
#pragma unroll
            for (int vv = 0; vv < 8; ++vv) {
                int off = 32 * s + 2 * vv + 16 * half;
                load_pair(kp + off, aK, vv);
            }
            acc = __builtin_amdgcn_wmma_f32_16x16x32_f16(
                false, aK, false, qf[s], (short)0, acc, false, false);
        }
        if (MASKED) {
#pragma unroll
            for (int r = 0; r < 8; ++r) {
                int j = jbase + 16 * ct + r + 8 * half;  // key of C' row r
                bool valid = (j <= qrow) && (j > qrow - WIN);
                sc[ct][r] = valid ? acc[r] : -1e30f;
            }
        } else {
#pragma unroll
            for (int r = 0; r < 8; ++r) sc[ct][r] = acc[r];
        }
    }

    // ---- per-lane online softmax (own row); halves combined with one xor-16 ----
    float lmax = sc[0][0];
#pragma unroll
    for (int ct = 0; ct < 2; ++ct)
#pragma unroll
        for (int r = 0; r < 8; ++r) lmax = fmaxf(lmax, sc[ct][r]);
    lmax = fmaxf(lmax, __shfl_xor(lmax, 16, 32));
    const float mold = mst;
    const float mnew = fmaxf(mold, lmax);
    const float corr = __builtin_amdgcn_exp2f(mold - mnew);
    mst = mnew;
    float p[2][8];
    float lsum = 0.0f;
#pragma unroll
    for (int ct = 0; ct < 2; ++ct)
#pragma unroll
        for (int r = 0; r < 8; ++r) {
            float pv;
            if (MASKED)
                pv = (sc[ct][r] > -1e29f) ? __builtin_amdgcn_exp2f(sc[ct][r] - mnew) : 0.0f;
            else
                pv = __builtin_amdgcn_exp2f(sc[ct][r] - mnew);
            p[ct][r] = pv;
            lsum += pv;
        }
    lsum += __shfl_xor(lsum, 16, 32);
    lst = lst * corr + lsum;
    // rescale accumulators only if some row's max advanced (wave-uniform branch)
    if (__ballot(mnew > mold)) {
#pragma unroll
        for (int dt = 0; dt < 8; ++dt)
#pragma unroll
            for (int r = 0; r < 8; ++r) oacc[dt][r] *= corr;
    }

    // ---- P^T -> B-fragment: pack f16 pairs, swap keys 8-15 <-> 16-23 across halves ----
    // half0 owns keys 0-7 (ct0) & 16-23 (ct1); half1 owns 8-15 & 24-31.
    // B layout needs half0: K=0..15, half1: K=16..31.
    union { v16h v; unsigned u[8]; } bp;
    unsigned own0[4], own1[4];
#pragma unroll
    for (int i = 0; i < 4; ++i) {
        union { unsigned u; _Float16 h[2]; } t0, t1;
        t0.h[0] = (_Float16)p[0][2 * i]; t0.h[1] = (_Float16)p[0][2 * i + 1];
        t1.h[0] = (_Float16)p[1][2 * i]; t1.h[1] = (_Float16)p[1][2 * i + 1];
        own0[i] = t0.u; own1[i] = t1.u;
    }
#pragma unroll
    for (int i = 0; i < 4; ++i) {
        unsigned snd = half ? own0[i] : own1[i];
        unsigned rcv = (unsigned)__shfl_xor((int)snd, 16, 32);
        bp.u[i]     = half ? rcv     : own0[i];
        bp.u[4 + i] = half ? own1[i] : rcv;
    }

    // ---- O^T += V^T x P^T : A = V^T (lane = d-row), B = P^T (lane = own q row) ----
#pragma unroll
    for (int dt = 0; dt < 8; ++dt) {
        v16h aV;
        const _Float16* vp = &Vb[(16 * dt + ln) * VSTR];
#pragma unroll
        for (int vv = 0; vv < 8; ++vv) {
            int off = 2 * vv + 16 * half;  // key index
            load_pair(vp + off, aV, vv);
        }
        oacc[dt] = __builtin_amdgcn_wmma_f32_16x16x32_f16(
            false, aV, false, bp.v, (short)0, oacc[dt], false, false);
    }
}

__global__ __launch_bounds__(256) void zSparseAttention_34746285425375_kernel(
    const float* __restrict__ q, const float* __restrict__ k,
    const float* __restrict__ v, float* __restrict__ out) {
    __shared__ __align__(16) _Float16 Ksh[2][32 * KSTR];      // K chunk (key, d), double-buffered
    __shared__ __align__(16) _Float16 Vtsh[2][DHEAD * VSTR];  // V chunk transposed (d, key)

    const int tid  = threadIdx.x;
    const int wave = tid >> 5;
    const int lane = tid & 31;
    const int ln   = lane & 15;
    const int half = lane >> 4;

    const int h    = blockIdx.x >> 5;  // 32 query-blocks of 128 rows per head
    const int qblk = blockIdx.x & 31;
    const int q0   = qblk * 128;
    const int q0w  = q0 + wave * 16;   // this wave's 16 query rows
    const int qrow = q0w + ln;         // this lane's query row (both halves)

    const size_t headoff = (size_t)h * S_LEN * DHEAD;
    const float* qh = q + headoff;
    const float* kh = k + headoff;
    const float* vh = v + headoff;
    float*       oh = out + headoff;

    // ---- Q as B-fragments (lane = own row); fold softmax scale AND log2e ----
    const float scale = 0.08838834764831845f * 1.4426950408889634f;  // 1/sqrt(128)*log2(e)
    v16h qf[4];
    {
        const float* qr = qh + (size_t)qrow * DHEAD;
#pragma unroll
        for (int s = 0; s < 4; ++s)
#pragma unroll
            for (int vv = 0; vv < 8; ++vv) {
                int d = 32 * s + 2 * vv + 16 * half;  // B layout: K = 2v + 16*half
                float2 dd = *reinterpret_cast<const float2*>(qr + d);
                qf[s][2 * vv]     = (_Float16)(dd.x * scale);
                qf[s][2 * vv + 1] = (_Float16)(dd.y * scale);
            }
    }

    v8f oacc[8];
#pragma unroll
    for (int i = 0; i < 8; ++i)
#pragma unroll
        for (int e = 0; e < 8; ++e) oacc[i][e] = 0.0f;
    float mst = -1e30f, lst = 0.0f;

    const int jlo = q0 - (WIN - 1);
    const int c0  = (jlo > 0 ? jlo : 0) >> 5;
    const int c1  = (q0 + 127) >> 5;

    // ---- software pipeline: stage next chunk's K/V in VGPRs ----
    float4 kst[4], vst[4];
    auto stage = [&](int c) {
#pragma unroll
        for (int i = 0; i < 4; ++i) {
            int idx  = tid + (i << 8);  // float4 index 0..1023
            int flat = idx << 2;
            int row  = flat >> 7;       // key within chunk 0..31
            int col  = flat & 127;      // d 0..124 step 4
            size_t g = (size_t)((c << 5) + row) * DHEAD + col;
            kst[i] = *reinterpret_cast<const float4*>(kh + g);
            vst[i] = *reinterpret_cast<const float4*>(vh + g);
        }
    };
    auto commit = [&](int p) {  // f32 -> f16, staged regs -> LDS buffer p
#pragma unroll
        for (int i = 0; i < 4; ++i) {
            int idx  = tid + (i << 8);
            int flat = idx << 2;
            int row  = flat >> 7;
            int col  = flat & 127;
            _Float16* kp = &Ksh[p][row * KSTR + col];
            kp[0] = (_Float16)kst[i].x; kp[1] = (_Float16)kst[i].y;
            kp[2] = (_Float16)kst[i].z; kp[3] = (_Float16)kst[i].w;
            _Float16* vp = &Vtsh[p][col * VSTR + row];
            vp[0 * VSTR] = (_Float16)vst[i].x;
            vp[1 * VSTR] = (_Float16)vst[i].y;
            vp[2 * VSTR] = (_Float16)vst[i].z;
            vp[3 * VSTR] = (_Float16)vst[i].w;
        }
    };

    stage(c0);
    for (int c = c0; c <= c1; ++c) {
        const int p     = c & 1;
        const int jbase = c << 5;
        commit(p);                 // waits staged loads; LDS[p] safe: barrier(c-1) passed
        if (c < c1) stage(c + 1);  // issue next chunk's global loads (latency hidden)
        __syncthreads();           // LDS[p] visible; prior reads of LDS[p^1] fenced next iter

        // wave-uniform band intersection / full-validity tests
        if ((jbase <= q0w + 15) && (jbase + 31 >= q0w - (WIN - 1))) {
            if ((jbase + 31 <= q0w) && (jbase >= q0w - (WIN - 16))) {
                chunk_compute<false>(Ksh[p], Vtsh[p], qf, oacc, mst, lst,
                                     jbase, qrow, ln, half);
            } else {
                chunk_compute<true>(Ksh[p], Vtsh[p], qf, oacc, mst, lst,
                                    jbase, qrow, ln, half);
            }
        }
    }

    // ---- epilogue: O^T lane = own row; d = 16*dt + r + 8*half contiguous per lane ----
    const float inv = __builtin_amdgcn_rcpf(lst);
    float* orow = oh + (size_t)qrow * DHEAD + 8 * half;
#pragma unroll
    for (int dt = 0; dt < 8; ++dt) {
        float4 a, b;
        a.x = oacc[dt][0] * inv; a.y = oacc[dt][1] * inv;
        a.z = oacc[dt][2] * inv; a.w = oacc[dt][3] * inv;
        b.x = oacc[dt][4] * inv; b.y = oacc[dt][5] * inv;
        b.z = oacc[dt][6] * inv; b.w = oacc[dt][7] * inv;
        *reinterpret_cast<float4*>(orow + 16 * dt)     = a;
        *reinterpret_cast<float4*>(orow + 16 * dt + 4) = b;
    }
}

extern "C" void kernel_launch(void* const* d_in, const int* in_sizes, int n_in,
                              void* d_out, int out_size, void* d_ws, size_t ws_size,
                              hipStream_t stream) {
    (void)in_sizes; (void)n_in; (void)out_size; (void)d_ws; (void)ws_size;
    const float* q = (const float*)d_in[0];
    const float* k = (const float*)d_in[1];
    const float* v = (const float*)d_in[2];
    float* out = (float*)d_out;
    // H=32 heads * 32 query-blocks of 128 rows = 1024 workgroups, 256 threads (8 wave32)
    dim3 grid(32 * 32);
    dim3 block(256);
    zSparseAttention_34746285425375_kernel<<<grid, block, 0, stream>>>(q, k, v, out);
}
